// MoEEP_4621384810814
// MI455X (gfx1250) — matrix operand order repaired and checked by reading, not compiled, not run on hardware
//
#include <hip/hip_runtime.h>
#include <math.h>

typedef __attribute__((ext_vector_type(16))) _Float16 v16h;
typedef __attribute__((ext_vector_type(8)))  _Float16 v8h;
typedef __attribute__((ext_vector_type(4)))  _Float16 v4h;
typedef __attribute__((ext_vector_type(8)))  float    v8f;
typedef __attribute__((ext_vector_type(4)))  float    v4f;

#define N_TOK 4096
#define DIM   1024
#define FEXP  1024
#define NEXP  8
#define MTILE 32          // gathered tokens per block
#define LDS_STRIDE 1032   // 1024 + 8 halves padding -> conflict-free row access

// workspace layout (bytes)
#define WS_COUNTS 0
#define WS_TOK    256
#define WS_WTS    (WS_TOK + NEXP * N_TOK * 4)
#define WS_W1H    (WS_WTS + NEXP * N_TOK * 4)
#define WS_W2H    (WS_W1H + (size_t)NEXP * FEXP * DIM * 2)
// total = ~33.8 MB

static __device__ __forceinline__ v16h cat16(v8h lo, v8h hi) {
  return __builtin_shufflevector(lo, hi, 0, 1, 2, 3, 4, 5, 6, 7,
                                 8, 9, 10, 11, 12, 13, 14, 15);
}

// Branch-free exact-GELU: erf via Abramowitz-Stegun 7.1.26 (|err| < 1.5e-7),
// raw v_exp_f32 / v_rcp_f32, no exec-mask divergence.
static __device__ __forceinline__ float gelu_f(float v) {
  float u = fabsf(v) * 0.70710678118654752f;          // |v|/sqrt(2)
  float t = __builtin_amdgcn_rcpf(1.0f + 0.3275911f * u);
  float p = t * (0.254829592f +
           t * (-0.284496736f +
           t * (1.421413741f +
           t * (-1.453152027f +
           t * 1.061405429f))));
  float er = 1.0f - p * __expf(-u * u);
  er = copysignf(er, v);
  return 0.5f * v * (1.0f + er);
}

// ---------------------------------------------------------------- zero kernel
__global__ __launch_bounds__(256) void moe_zero_kernel(float* __restrict__ out,
                                                       int n,
                                                       int* __restrict__ counts) {
  int i = blockIdx.x * 256 + threadIdx.x;
  if (i < n) out[i] = 0.0f;
  if (blockIdx.x == 0 && threadIdx.x < NEXP) counts[threadIdx.x] = 0;
}

// -------------------------------------------------------- weight f32->f16 cvt
__global__ __launch_bounds__(256) void moe_cvt_kernel(const float* __restrict__ w,
                                                      _Float16* __restrict__ wh,
                                                      int n4) {
  int i = blockIdx.x * 256 + threadIdx.x;
  if (i < n4) {
    v4f f = *reinterpret_cast<const v4f*>(w + (size_t)i * 4);
    v4h h;
    h.x = (_Float16)f.x; h.y = (_Float16)f.y;
    h.z = (_Float16)f.z; h.w = (_Float16)f.w;
    *reinterpret_cast<v4h*>(wh + (size_t)i * 4) = h;
  }
}

// ------------------------------------------------------- router + dispatch
__global__ __launch_bounds__(256) void moe_router_kernel(
    const float* __restrict__ x, const float* __restrict__ rw,
    const float* __restrict__ bias, int* __restrict__ counts,
    int* __restrict__ tok, float* __restrict__ wts) {
  const int wave = threadIdx.x >> 5;
  const int lane = threadIdx.x & 31;
  const int t = blockIdx.x * 8 + wave;   // one wave per token

  float acc[NEXP];
#pragma unroll
  for (int e = 0; e < NEXP; ++e) acc[e] = 0.0f;

  const float* xr = x + (size_t)t * DIM;
  for (int d = lane; d < DIM; d += 32) {
    float xv = xr[d];
#pragma unroll
    for (int e = 0; e < NEXP; ++e) acc[e] += xv * rw[e * DIM + d];
  }
#pragma unroll
  for (int off = 16; off > 0; off >>= 1) {
#pragma unroll
    for (int e = 0; e < NEXP; ++e) acc[e] += __shfl_down(acc[e], off);
  }

  if (lane == 0) {
    // top-2 on bias-adjusted logits (first index wins ties, matching top_k)
    int i0 = 0;
    float b0 = acc[0] + bias[0];
    for (int e = 1; e < NEXP; ++e) {
      float b = acc[e] + bias[e];
      if (b > b0) { b0 = b; i0 = e; }
    }
    int i1 = -1;
    float b1 = -3.402823466e+38f;
    for (int e = 0; e < NEXP; ++e) {
      if (e == i0) continue;
      float b = acc[e] + bias[e];
      if (b > b1) { b1 = b; i1 = e; }
    }
    // softmax on the RAW logits of the selected pair
    float l0 = acc[i0], l1 = acc[i1];
    float m = fmaxf(l0, l1);
    float e0 = __expf(l0 - m), e1 = __expf(l1 - m);
    float inv = __builtin_amdgcn_rcpf(e0 + e1);

    int p0 = atomicAdd(&counts[i0], 1);
    tok[i0 * N_TOK + p0] = t;
    wts[i0 * N_TOK + p0] = e0 * inv;
    int p1 = atomicAdd(&counts[i1], 1);
    tok[i1 * N_TOK + p1] = t;
    wts[i1 * N_TOK + p1] = e1 * inv;
  }
}

// --------------------------------------------------- fused expert FFN kernel
// grid: (N_TOK/MTILE tiles, NEXP experts), 256 threads = 8 waves.
// Tile = 32 gathered tokens (two 16-row WMMA m-tiles share each B fragment).
// Each wave owns 128 output columns (8 16x16 n-tiles).
__global__ __launch_bounds__(256) void moe_expert_kernel(
    const float* __restrict__ x, const _Float16* __restrict__ w1h,
    const _Float16* __restrict__ w2h, const int* __restrict__ counts,
    const int* __restrict__ tok, const float* __restrict__ wts,
    float* __restrict__ out) {
  __shared__ _Float16 ldsX[MTILE * LDS_STRIDE];
  __shared__ _Float16 ldsH[MTILE * LDS_STRIDE];
  __shared__ int sTok[MTILE];
  __shared__ float sWt[MTILE];

  const int e = blockIdx.y;
  const int m0 = blockIdx.x * MTILE;
  const int cnt = counts[e];
  if (m0 >= cnt) return;   // uniform exit: EXEC stays all-ones for WMMA

  const int tid = threadIdx.x;
  if (tid < MTILE) {
    int r = m0 + tid;
    sTok[tid] = (r < cnt) ? tok[e * N_TOK + r] : -1;
    sWt[tid]  = (r < cnt) ? wts[e * N_TOK + r] : 0.0f;
  }
  __syncthreads();

  // stage gathered X tile into LDS as f16 (zeros for tail rows)
  for (int idx = tid; idx < MTILE * DIM; idx += 256) {
    int r = idx >> 10;
    int d = idx & (DIM - 1);
    int t = sTok[r];
    float v = (t >= 0) ? x[(size_t)t * DIM + d] : 0.0f;
    ldsX[r * LDS_STRIDE + d] = (_Float16)v;
  }
  __syncthreads();

  const int wave = tid >> 5;
  const int lane = tid & 31;
  const int n  = lane & 15;   // A row index / B row index / C column
  const int hi = lane >> 4;   // K-half selector

  // ---------------- Phase 1: H = gelu(X @ W1^T), cols [wave*128, +128)
  for (int nt = 0; nt < 8; ++nt) {
    const int f0 = wave * 128 + nt * 16;
    const _Float16* brow = w1h + ((size_t)(e * FEXP + f0 + n)) * DIM;
    __builtin_prefetch(brow, 0, 1);
    v8f acc0 = {};
    v8f acc1 = {};
#pragma unroll 2
    for (int k = 0; k < DIM; k += 32) {
      v8h b0 = *reinterpret_cast<const v8h*>(brow + k + hi * 8);
      v8h b1 = *reinterpret_cast<const v8h*>(brow + k + 16 + hi * 8);
      v16h B = cat16(b0, b1);
      v8h a0 = *reinterpret_cast<const v8h*>(&ldsX[n * LDS_STRIDE + k + hi * 8]);
      v8h a1 = *reinterpret_cast<const v8h*>(&ldsX[n * LDS_STRIDE + k + 16 + hi * 8]);
      acc0 = __builtin_amdgcn_wmma_f32_16x16x32_f16(
          false, cat16(a0, a1), false, B, (short)0, acc0, false, false);
      v8h a2 = *reinterpret_cast<const v8h*>(&ldsX[(16 + n) * LDS_STRIDE + k + hi * 8]);
      v8h a3 = *reinterpret_cast<const v8h*>(&ldsX[(16 + n) * LDS_STRIDE + k + 16 + hi * 8]);
      acc1 = __builtin_amdgcn_wmma_f32_16x16x32_f16(
          false, cat16(a2, a3), false, B, (short)0, acc1, false, false);
    }
#pragma unroll
    for (int j = 0; j < 8; ++j) {
      ldsH[(hi * 8 + j) * LDS_STRIDE + f0 + n]        = (_Float16)gelu_f(acc0[j]);
      ldsH[(16 + hi * 8 + j) * LDS_STRIDE + f0 + n]   = (_Float16)gelu_f(acc1[j]);
    }
  }
  __syncthreads();

  // ---------------- Phase 2: Y = H @ W2^T, scale by combine wt, scatter-add
  for (int nt = 0; nt < 8; ++nt) {
    const int d0 = wave * 128 + nt * 16;
    const _Float16* brow = w2h + ((size_t)(e * DIM + d0 + n)) * FEXP;
    __builtin_prefetch(brow, 0, 1);
    v8f acc0 = {};
    v8f acc1 = {};
#pragma unroll 2
    for (int k = 0; k < FEXP; k += 32) {
      v8h b0 = *reinterpret_cast<const v8h*>(brow + k + hi * 8);
      v8h b1 = *reinterpret_cast<const v8h*>(brow + k + 16 + hi * 8);
      v16h B = cat16(b0, b1);
      v8h a0 = *reinterpret_cast<const v8h*>(&ldsH[n * LDS_STRIDE + k + hi * 8]);
      v8h a1 = *reinterpret_cast<const v8h*>(&ldsH[n * LDS_STRIDE + k + 16 + hi * 8]);
      acc0 = __builtin_amdgcn_wmma_f32_16x16x32_f16(
          false, cat16(a0, a1), false, B, (short)0, acc0, false, false);
      v8h a2 = *reinterpret_cast<const v8h*>(&ldsH[(16 + n) * LDS_STRIDE + k + hi * 8]);
      v8h a3 = *reinterpret_cast<const v8h*>(&ldsH[(16 + n) * LDS_STRIDE + k + 16 + hi * 8]);
      acc1 = __builtin_amdgcn_wmma_f32_16x16x32_f16(
          false, cat16(a2, a3), false, B, (short)0, acc1, false, false);
    }
#pragma unroll
    for (int j = 0; j < 8; ++j) {
      int r0 = hi * 8 + j;
      int r1 = 16 + r0;
      int t0 = sTok[r0];
      int t1 = sTok[r1];
      if (t0 >= 0) atomicAdd(&out[(size_t)t0 * DIM + d0 + n], acc0[j] * sWt[r0]);
      if (t1 >= 0) atomicAdd(&out[(size_t)t1 * DIM + d0 + n], acc1[j] * sWt[r1]);
    }
  }
}

// ------------------------------------------------------------------ launcher
extern "C" void kernel_launch(void* const* d_in, const int* in_sizes, int n_in,
                              void* d_out, int out_size, void* d_ws, size_t ws_size,
                              hipStream_t stream) {
  (void)in_sizes; (void)n_in; (void)out_size; (void)ws_size;

  const float* x    = (const float*)d_in[0];  // [B,T,D] = [4096,1024]
  const float* bias = (const float*)d_in[1];  // [E]
  const float* rw   = (const float*)d_in[2];  // [E,D]
  const float* w1   = (const float*)d_in[3];  // [E,F,D]
  const float* w2   = (const float*)d_in[4];  // [E,D,F]
  float* out = (float*)d_out;

  char* ws = (char*)d_ws;
  int*       counts = (int*)(ws + WS_COUNTS);
  int*       tok    = (int*)(ws + WS_TOK);
  float*     wts    = (float*)(ws + WS_WTS);
  _Float16*  w1h    = (_Float16*)(ws + WS_W1H);
  _Float16*  w2h    = (_Float16*)(ws + WS_W2H);

  // 1) zero output + expert counts
  {
    int n = N_TOK * DIM;
    moe_zero_kernel<<<(n + 255) / 256, 256, 0, stream>>>(out, n, counts);
  }
  // 2) convert weights to f16 workspace
  {
    int n4 = (NEXP * FEXP * DIM) / 4;
    moe_cvt_kernel<<<(n4 + 255) / 256, 256, 0, stream>>>(w1, w1h, n4);
    moe_cvt_kernel<<<(n4 + 255) / 256, 256, 0, stream>>>(w2, w2h, n4);
  }
  // 3) route + dispatch (one wave per token, 8 tokens per block)
  moe_router_kernel<<<N_TOK / 8, 256, 0, stream>>>(x, rw, bias, counts, tok, wts);
  // 4) fused expert FFN over gathered tiles
  {
    dim3 grid(N_TOK / MTILE, NEXP);
    moe_expert_kernel<<<grid, 256, 0, stream>>>(x, w1h, w2h, counts, tok, wts, out);
  }
}
